// LSTM_Model_15805479649304
// MI455X (gfx1250) — matrix operand order repaired and checked by reading, not compiled, not run on hardware
//
#include <hip/hip_runtime.h>

#define H  1024
#define H3 3072
#define KTILE 32

typedef __bf16 v16bf __attribute__((ext_vector_type(16)));
typedef float  v8f   __attribute__((ext_vector_type(8)));

union Frag { v16bf v; uint4 q[2]; };

static __device__ __forceinline__ float bf2f(unsigned short u) {
    return __uint_as_float(((unsigned int)u) << 16);
}
static __device__ __forceinline__ unsigned short f2bf(float f) {
    unsigned int u = __float_as_uint(f);
    u += 0x7FFFu + ((u >> 16) & 1u);
    return (unsigned short)(u >> 16);
}
static __device__ __forceinline__ float sigf(float x) {
    return 1.0f / (1.0f + __expf(-x));
}
// LDS aperture keeps the byte offset in addr[31:0]; truncate generic pointer.
static __device__ __forceinline__ unsigned ldsaddr(const void* p) {
    return (unsigned)(size_t)p;
}
// CDNA5 async DMA: global -> LDS, tracked by ASYNCcnt (no VGPR round trip).
static __device__ __forceinline__ void async_b128(unsigned lds, const void* g) {
    asm volatile("global_load_async_to_lds_b128 %0, %1, off"
                 :: "v"(lds), "v"((unsigned long long)(size_t)g) : "memory");
}

// ---------------- prep kernels ----------------

__global__ void kconv(const float* __restrict__ src, unsigned short* __restrict__ dst, int n) {
    int i = blockIdx.x * blockDim.x + threadIdx.x;
    if (i < n) dst[i] = f2bf(src[i]);
}

// stage means of broadcast weather folded into layer0 input-path constants:
// gw_all[t][row] = bih0[row] + sum_c wmean[t][c]*Wih0[row][c] ; wv8[row] = Wih0[row][7]
__global__ void kweather(const float* __restrict__ w, const float* __restrict__ Wih0,
                         const float* __restrict__ bih0,
                         float* __restrict__ gw_all, float* __restrict__ wv8) {
    __shared__ float wm[4][7];
    const int tid = threadIdx.x;
    const int a_[4] = {0, 13, 18, 22};
    const int b_[4] = {13, 18, 22, 26};
    if (tid < 28) {
        int st = tid / 7, c = tid % 7;
        float s = 0.f;
        for (int k = a_[st]; k < b_[st]; ++k) s += w[c * 26 + k];
        wm[st][c] = s / (float)(b_[st] - a_[st]);
    }
    __syncthreads();
    for (int row = tid; row < H3; row += blockDim.x) {
        wv8[row] = Wih0[row * 8 + 7];
        float bias = bih0[row];
        for (int st = 0; st < 4; ++st) {
            float acc = bias;
            #pragma unroll
            for (int c = 0; c < 7; ++c) acc += wm[st][c] * Wih0[row * 8 + c];
            gw_all[st * H3 + row] = acc;
        }
    }
}

// soil_h0 = tanh(soil[:, :11] @ W_rs^T + b_rs) -> bf16 init for both layers
__global__ void ksoil(const float* __restrict__ soil, const float* __restrict__ Wrs,
                      const float* __restrict__ brs,
                      unsigned short* __restrict__ h0b, unsigned short* __restrict__ h1b) {
    __shared__ float s[11];
    const int b = blockIdx.x, tid = threadIdx.x;
    if (tid < 11) s[tid] = soil[(size_t)b * 13 + tid];
    __syncthreads();
    for (int jj = 0; jj < H / 256; ++jj) {
        int j = jj * 256 + tid;
        float acc = brs[j];
        #pragma unroll
        for (int k = 0; k < 11; ++k) acc += s[k] * Wrs[j * 11 + k];
        unsigned short v = f2bf(tanhf(acc));
        h0b[(size_t)b * H + j] = v;
        h1b[(size_t)b * H + j] = v;
    }
}

// ---------------- fused GRU cell kernels (bf16 WMMA, async-LDS pipelined) -----
// Workgroup tile: 64 batch rows x 64 hidden cols; 8 waves, each a 32x16 output
// (B-fragments reused across 2 M-subtiles). Double-buffered LDS filled by
// GLOBAL_LOAD_ASYNC_TO_LDS_B128; each thread issues a wave-uniform load count
// so a single s_wait_asynccnt immediate retires exactly one stage.

__global__ __launch_bounds__(256) void gru_layer0(
    const unsigned short* __restrict__ hin,   // [B,H] bf16
    const unsigned short* __restrict__ Whh,   // [3H,H] bf16
    const float* __restrict__ gw,             // [3H] (weather + bih0 folded)
    const float* __restrict__ wv8,            // [3H] prev-yield column of Wih0
    const float* __restrict__ bhh,            // [3H]
    const float* __restrict__ oprev,          // [B] or nullptr
    unsigned short* __restrict__ hout) {
    __shared__ __align__(16) unsigned short hs[2][64][40];
    __shared__ __align__(16) unsigned short wsh[2][3][64][40];
    const int tid  = threadIdx.x;
    const int lane = tid & 31, wave = tid >> 5;
    const int wm = wave & 1, wn = wave >> 1;   // 2 x 4 wave grid
    const int jbase = blockIdx.x * 64;
    const int bbase = blockIdx.y * 64;
    const int mrow = lane & 15, hiK = lane >> 4;
    const int lr = tid >> 2, lco = (tid & 3) * 8;

    v8f zero = {};
    v8f acc[6];                                // 3 gates x 2 M-subtiles
    #pragma unroll
    for (int g = 0; g < 6; ++g) acc[g] = zero;

    const unsigned short* hsrc = hin + (size_t)(bbase + lr) * H + lco;
    const unsigned short* wsrc[3];
    #pragma unroll
    for (int g = 0; g < 3; ++g)
        wsrc[g] = Whh + ((size_t)g * H + jbase + lr) * H + lco;

    auto issue = [&](int s, int kc) {          // 4 async b128 per thread
        async_b128(ldsaddr(&hs[s][lr][lco]), hsrc + kc);
        #pragma unroll
        for (int g = 0; g < 3; ++g)
            async_b128(ldsaddr(&wsh[s][g][lr][lco]), wsrc[g] + kc);
    };

    issue(0, 0);
    for (int it = 0; it < H / KTILE; ++it) {
        const int s = it & 1;
        __syncthreads();                       // readers of buffer s^1 done
        if (it + 1 < H / KTILE) {
            issue(s ^ 1, (it + 1) * KTILE);
            asm volatile("s_wait_asynccnt 0x4" ::: "memory");  // stage s complete
        } else {
            asm volatile("s_wait_asynccnt 0x0" ::: "memory");
        }
        __syncthreads();                       // whole tile visible

        Frag ah[2];
        #pragma unroll
        for (int ms = 0; ms < 2; ++ms) {
            const int row = wm * 32 + ms * 16 + mrow;
            ah[ms].q[0] = *(const uint4*)&hs[s][row][hiK * 8];
            ah[ms].q[1] = *(const uint4*)&hs[s][row][16 + hiK * 8];
        }
        #pragma unroll
        for (int g = 0; g < 3; ++g) {
            Frag bw;
            bw.q[0] = *(const uint4*)&wsh[s][g][wn * 16 + mrow][hiK * 8];
            bw.q[1] = *(const uint4*)&wsh[s][g][wn * 16 + mrow][16 + hiK * 8];
            #pragma unroll
            for (int ms = 0; ms < 2; ++ms)
                acc[g * 2 + ms] = __builtin_amdgcn_wmma_f32_16x16x32_bf16(
                    false, ah[ms].v, false, bw.v, (short)0, acc[g * 2 + ms], false, false);
        }
    }

    const int j = jbase + wn * 16 + mrow;
    const float bhr = bhh[j], bhz = bhh[H + j], bhn = bhh[2 * H + j];
    const float gwr = gw[j],  gwz = gw[H + j],  gwn = gw[2 * H + j];
    const float wvr = wv8[j], wvz = wv8[H + j], wvn = wv8[2 * H + j];
    #pragma unroll
    for (int ms = 0; ms < 2; ++ms) {
        const int brow0 = bbase + wm * 32 + ms * 16 + hiK * 8;
        #pragma unroll
        for (int q = 0; q < 8; ++q) {
            int b = brow0 + q;
            float op = oprev ? oprev[b] : 0.f;
            float r = sigf(gwr + op * wvr + acc[0 + ms][q] + bhr);
            float z = sigf(gwz + op * wvz + acc[2 + ms][q] + bhz);
            float n = tanhf(gwn + op * wvn + r * (acc[4 + ms][q] + bhn));
            float hp = bf2f(hin[(size_t)b * H + j]);
            hout[(size_t)b * H + j] = f2bf((1.f - z) * n + z * hp);
        }
    }
}

__global__ __launch_bounds__(256) void gru_layer1(
    const unsigned short* __restrict__ xin,   // [B,H] bf16 (= new h0)
    const unsigned short* __restrict__ hin,   // [B,H] bf16
    const unsigned short* __restrict__ Wih,   // [3H,H] bf16
    const unsigned short* __restrict__ Whh,   // [3H,H] bf16
    const float* __restrict__ bih, const float* __restrict__ bhh,
    unsigned short* __restrict__ hout) {
    __shared__ __align__(16) unsigned short xs[2][64][40];
    __shared__ __align__(16) unsigned short hs[2][64][40];
    __shared__ __align__(16) unsigned short wsh[2][6][64][40];
    const int tid  = threadIdx.x;
    const int lane = tid & 31, wave = tid >> 5;
    const int wm = wave & 1, wn = wave >> 1;
    const int jbase = blockIdx.x * 64;
    const int bbase = blockIdx.y * 64;
    const int mrow = lane & 15, hiK = lane >> 4;
    const int lr = tid >> 2, lco = (tid & 3) * 8;

    v8f zero = {};
    v8f acc[12];                               // (gi r,z,n | gh r,z,n) x 2 M-subtiles
    #pragma unroll
    for (int g = 0; g < 12; ++g) acc[g] = zero;

    const unsigned short* xsrc = xin + (size_t)(bbase + lr) * H + lco;
    const unsigned short* hsrc = hin + (size_t)(bbase + lr) * H + lco;
    const unsigned short* wsrc[6];
    #pragma unroll
    for (int g = 0; g < 3; ++g) {
        wsrc[g]     = Wih + ((size_t)g * H + jbase + lr) * H + lco;
        wsrc[g + 3] = Whh + ((size_t)g * H + jbase + lr) * H + lco;
    }

    auto issue = [&](int s, int kc) {          // 8 async b128 per thread
        async_b128(ldsaddr(&xs[s][lr][lco]), xsrc + kc);
        async_b128(ldsaddr(&hs[s][lr][lco]), hsrc + kc);
        #pragma unroll
        for (int g = 0; g < 6; ++g)
            async_b128(ldsaddr(&wsh[s][g][lr][lco]), wsrc[g] + kc);
    };

    issue(0, 0);
    for (int it = 0; it < H / KTILE; ++it) {
        const int s = it & 1;
        __syncthreads();
        if (it + 1 < H / KTILE) {
            issue(s ^ 1, (it + 1) * KTILE);
            asm volatile("s_wait_asynccnt 0x8" ::: "memory");
        } else {
            asm volatile("s_wait_asynccnt 0x0" ::: "memory");
        }
        __syncthreads();

        Frag ax[2], ah[2];
        #pragma unroll
        for (int ms = 0; ms < 2; ++ms) {
            const int row = wm * 32 + ms * 16 + mrow;
            ax[ms].q[0] = *(const uint4*)&xs[s][row][hiK * 8];
            ax[ms].q[1] = *(const uint4*)&xs[s][row][16 + hiK * 8];
            ah[ms].q[0] = *(const uint4*)&hs[s][row][hiK * 8];
            ah[ms].q[1] = *(const uint4*)&hs[s][row][16 + hiK * 8];
        }
        #pragma unroll
        for (int g = 0; g < 6; ++g) {
            Frag bw;
            bw.q[0] = *(const uint4*)&wsh[s][g][wn * 16 + mrow][hiK * 8];
            bw.q[1] = *(const uint4*)&wsh[s][g][wn * 16 + mrow][16 + hiK * 8];
            #pragma unroll
            for (int ms = 0; ms < 2; ++ms)
                acc[g * 2 + ms] = __builtin_amdgcn_wmma_f32_16x16x32_bf16(
                    false, (g < 3) ? ax[ms].v : ah[ms].v, false, bw.v,
                    (short)0, acc[g * 2 + ms], false, false);
        }
    }

    const int j = jbase + wn * 16 + mrow;
    const float bir = bih[j], biz = bih[H + j], bin_ = bih[2 * H + j];
    const float bhr = bhh[j], bhz = bhh[H + j], bhn  = bhh[2 * H + j];
    #pragma unroll
    for (int ms = 0; ms < 2; ++ms) {
        const int brow0 = bbase + wm * 32 + ms * 16 + hiK * 8;
        #pragma unroll
        for (int q = 0; q < 8; ++q) {
            int b = brow0 + q;
            float r = sigf(acc[0 + ms][q] + bir + acc[6 + ms][q] + bhr);
            float z = sigf(acc[2 + ms][q] + biz + acc[8 + ms][q] + bhz);
            float n = tanhf(acc[4 + ms][q] + bin_ + r * (acc[10 + ms][q] + bhn));
            float hp = bf2f(hin[(size_t)b * H + j]);
            hout[(size_t)b * H + j] = f2bf((1.f - z) * n + z * hp);
        }
    }
}

// ---------------- heads ----------------

// o[b] = tanh(h1[b,:] . Wfc2 + b) ; one wave32 per row
__global__ void kfc2(const unsigned short* __restrict__ h1,
                     const float* __restrict__ Wfc2, const float* __restrict__ bfc2,
                     float* __restrict__ out, int B_) {
    const int lane = threadIdx.x & 31;
    const int wv   = threadIdx.x >> 5;
    const int b = blockIdx.x * (blockDim.x >> 5) + wv;
    if (b >= B_) return;
    float acc = 0.f;
    #pragma unroll 4
    for (int i = 0; i < H / 32; ++i) {
        int c = i * 32 + lane;
        acc += bf2f(h1[(size_t)b * H + c]) * Wfc2[c];
    }
    #pragma unroll
    for (int m = 16; m; m >>= 1) acc += __shfl_xor(acc, m, 32);
    if (lane == 0) out[b] = tanhf(acc + bfc2[0]);
}

__global__ void kfinal(const float* __restrict__ soil, const float* __restrict__ Wsf,
                       const float* __restrict__ bsf, const float* __restrict__ Wy3,
                       const float* __restrict__ by3, float* __restrict__ out, int B_) {
    const int b = blockIdx.x * blockDim.x + threadIdx.x;
    if (b >= B_) return;
    float acc = bsf[0];
    #pragma unroll
    for (int k = 0; k < 11; ++k) acc += soil[(size_t)b * 13 + k] * Wsf[k];
    float xgb = tanhf(acc);
    float f = by3[0] + xgb * Wy3[4];
    #pragma unroll
    for (int s = 0; s < 4; ++s) f += out[(size_t)s * B_ + b] * Wy3[s];
    out[(size_t)4 * B_ + b] = f;
}

// ---------------- host ----------------

extern "C" void kernel_launch(void* const* d_in, const int* in_sizes, int n_in,
                              void* d_out, int out_size, void* d_ws, size_t ws_size,
                              hipStream_t stream) {
    const float* soil     = (const float*)d_in[0];
    const float* weather1 = (const float*)d_in[1];
    const float* Wrs  = (const float*)d_in[6];
    const float* brs  = (const float*)d_in[7];
    const float* Wfc2 = (const float*)d_in[8];
    const float* bfc2 = (const float*)d_in[9];
    const float* Wsf  = (const float*)d_in[10];
    const float* bsf  = (const float*)d_in[11];
    const float* Wy3  = (const float*)d_in[12];
    const float* by3  = (const float*)d_in[13];
    const float* Wih0 = (const float*)d_in[14];
    const float* Whh0 = (const float*)d_in[15];
    const float* bih0 = (const float*)d_in[16];
    const float* bhh0 = (const float*)d_in[17];
    const float* Wih1 = (const float*)d_in[18];
    const float* Whh1 = (const float*)d_in[19];
    const float* bih1 = (const float*)d_in[20];
    const float* bhh1 = (const float*)d_in[21];

    const int B_ = in_sizes[0] / 13;
    float* out = (float*)d_out;

    char* ws = (char*)d_ws;
    size_t off = 0;
    auto alloc = [&](size_t bytes) -> void* {
        void* p = ws + off;
        off += (bytes + 255) & ~(size_t)255;
        return p;
    };
    unsigned short* Whh0b  = (unsigned short*)alloc((size_t)H3 * H * 2);
    unsigned short* Wih1b  = (unsigned short*)alloc((size_t)H3 * H * 2);
    unsigned short* Whh1b  = (unsigned short*)alloc((size_t)H3 * H * 2);
    unsigned short* h0b[2] = {(unsigned short*)alloc((size_t)B_ * H * 2),
                              (unsigned short*)alloc((size_t)B_ * H * 2)};
    unsigned short* h1b[2] = {(unsigned short*)alloc((size_t)B_ * H * 2),
                              (unsigned short*)alloc((size_t)B_ * H * 2)};
    float* gw_all = (float*)alloc((size_t)4 * H3 * 4);
    float* wv8    = (float*)alloc((size_t)H3 * 4);
    (void)n_in; (void)out_size; (void)ws_size;

    const int nW = H3 * H;
    kconv<<<(nW + 255) / 256, 256, 0, stream>>>(Whh0, Whh0b, nW);
    kconv<<<(nW + 255) / 256, 256, 0, stream>>>(Wih1, Wih1b, nW);
    kconv<<<(nW + 255) / 256, 256, 0, stream>>>(Whh1, Whh1b, nW);
    kweather<<<1, 256, 0, stream>>>(weather1, Wih0, bih0, gw_all, wv8);
    ksoil<<<B_, 256, 0, stream>>>(soil, Wrs, brs, h0b[0], h1b[0]);

    dim3 ggrid(H / 64, B_ / 64);
    int c0 = 0, c1 = 0;
    // 5 rnn calls: t indices 0,1,2,3,3 ; call3's output is overwritten -> skip head
    const int tmap[5] = {0, 1, 2, 3, 3};
    const int slot[5] = {0, 1, 2, -1, 3};
    const float* oprev = nullptr;
    for (int call = 0; call < 5; ++call) {
        gru_layer0<<<ggrid, 256, 0, stream>>>(h0b[c0], Whh0b, gw_all + tmap[call] * H3,
                                              wv8, bhh0, oprev, h0b[c0 ^ 1]);
        c0 ^= 1;
        gru_layer1<<<ggrid, 256, 0, stream>>>(h0b[c0], h1b[c1], Wih1b, Whh1b,
                                              bih1, bhh1, h1b[c1 ^ 1]);
        c1 ^= 1;
        if (slot[call] >= 0)
            kfc2<<<B_ / 4, 128, 0, stream>>>(h1b[c1], Wfc2, bfc2,
                                             out + (size_t)slot[call] * B_, B_);
        if (call == 0)      oprev = out;                    // o0
        else if (call == 1) oprev = out + (size_t)B_;       // o1
        else                oprev = out + 2 * (size_t)B_;   // o2 (calls 3 and 4)
    }
    kfinal<<<(B_ + 255) / 256, 256, 0, stream>>>(soil, Wsf, bsf, Wy3, by3, out, B_);
}
